// SwitchHeadCore_57698590655102
// MI455X (gfx1250) — matrix operand order repaired and checked by reading, not compiled, not run on hardware
//
#include <hip/hip_runtime.h>

// ---------------- types ----------------
typedef __attribute__((ext_vector_type(16))) __bf16 v16bf;
typedef __attribute__((ext_vector_type(8)))  __bf16 v8bf;
typedef __attribute__((ext_vector_type(8)))  float  v8f;

__device__ inline v8f zero8() {
  v8f z;
#pragma unroll
  for (int i = 0; i < 8; i++) z[i] = 0.f;
  return z;
}

__device__ inline v16bf make_v16(v8bf lo, v8bf hi) {
  v16bf r;
#pragma unroll
  for (int i = 0; i < 8; i++) { r[i] = lo[i]; r[i + 8] = hi[i]; }
  return r;
}

// A fragment: 16xK tile (row-major, leading dim ld), K-offset kb (multiple of 32).
// ISA layout: lanes0-15 row M=lane K=0..7 & 16..23 ; lanes16-31 same rows K=8..15 & 24..31.
__device__ inline v16bf load_a(const __bf16* base, int ld, int kb, int lane) {
  int row = lane & 15, kh = (lane >> 4) << 3;
  const __bf16* p = base + (long)row * ld + kb + kh;
  return make_v16(*(const v8bf*)p, *(const v8bf*)(p + 16));
}

// B fragment: Kx16 tile; memory layout has element (k,n) at base + n*ld + k (K contiguous).
// ISA layout: lanes0-15 col N=lane K=0..7 & 16..23 ; lanes16-31 same cols K=8..15 & 24..31.
__device__ inline v16bf load_b(const __bf16* base, int ld, int kb, int lane) {
  int col = lane & 15, kh = (lane >> 4) << 3;
  const __bf16* p = base + (long)col * ld + kb + kh;
  return make_v16(*(const v8bf*)p, *(const v8bf*)(p + 16));
}

__device__ inline v8f wmma_bf16(v16bf a, v16bf b, v8f c) {
  return __builtin_amdgcn_wmma_f32_16x16x32_bf16(false, a, false, b, (short)0, c, false, false);
}

// ---------------- fp32 -> bf16 elementwise ----------------
__global__ void cvt_bf16_kernel(const float* __restrict__ in, __bf16* __restrict__ out, long n) {
  long i = (long)blockIdx.x * blockDim.x + threadIdx.x;
  if (i < n) out[i] = (__bf16)in[i];
}

// ---------------- fp32 transpose (per z-slice rows x cols -> cols x rows) -> bf16 ----------------
__global__ void tcvt_kernel(const float* __restrict__ in, __bf16* __restrict__ out,
                            int rows, int cols) {
  __shared__ float tile[32][33];
  int z = blockIdx.z;
  const float* src = in + (long)z * rows * cols;
  __bf16* dst = out + (long)z * rows * cols;
  int r0 = blockIdx.y * 32, c0 = blockIdx.x * 32;
  int tx = threadIdx.x, ty = threadIdx.y;  // 32 x 8
  for (int i = 0; i < 32; i += 8) {
    int r = r0 + ty + i, c = c0 + tx;
    tile[ty + i][tx] = (r < rows && c < cols) ? src[(long)r * cols + c] : 0.f;
  }
  __syncthreads();
  for (int i = 0; i < 32; i += 8) {
    int r = c0 + ty + i, c = r0 + tx;  // output row = original col
    if (r < cols && c < rows) dst[(long)r * rows + c] = (__bf16)tile[tx][ty + i];
  }
}

// ---------------- gates: sigmoid(src @ selw.T) per head top-2, dense-masked ----------------
__global__ void gates_kernel(const float* __restrict__ src, const float* __restrict__ selw,
                             float* __restrict__ gdense) {
  __shared__ float s_src[1024];
  __shared__ float s_logit[64];
  int tok = blockIdx.x;   // 0 .. B*S-1
  int t = threadIdx.x;    // 0 .. 63
  const float* sp = src + (long)tok * 1024;
  for (int i = t; i < 1024; i += 64) s_src[i] = sp[i];
  __syncthreads();
  const float* w = selw + (long)t * 1024;
  float acc = 0.f;
  for (int i = 0; i < 1024; i += 4)
    acc += w[i] * s_src[i] + w[i + 1] * s_src[i + 1] + w[i + 2] * s_src[i + 2] + w[i + 3] * s_src[i + 3];
  s_logit[t] = acc;
  __syncthreads();
  if (t < 8) {  // head t: sigmoid + top-2 over 8 experts
    float g[8];
    for (int e = 0; e < 8; e++) g[e] = 1.f / (1.f + __expf(-s_logit[t * 8 + e]));
    int i1 = 0;
    for (int e = 1; e < 8; e++) if (g[e] > g[i1]) i1 = e;
    int i2 = -1;
    for (int e = 0; e < 8; e++) if (e != i1 && (i2 < 0 || g[e] > g[i2])) i2 = e;
    float* o = gdense + ((long)tok * 8 + t) * 8;
    for (int e = 0; e < 8; e++) o[e] = (e == i1 || e == i2) ? g[e] : 0.f;
  }
}

// ---------------- q/k projection: out[b,h,s,p] = scale * (src @ W^T), 32x32 tile/wave ----------------
__global__ void qkproj_kernel(const __bf16* __restrict__ A, const __bf16* __restrict__ W,
                              __bf16* __restrict__ out, float scale) {
  int wid = (blockIdx.x * blockDim.x + threadIdx.x) >> 5;
  int lane = threadIdx.x & 31;
  int nt = wid & 31, mt = wid >> 5;       // 32 n-tiles, 128 m-tiles (each 32 wide)
  int m0 = mt * 32, n0 = nt * 32;
  const __bf16* Ab = A + (long)m0 * 1024;
  const __bf16* Wb = W + (long)n0 * 1024;  // column n == row n of W, K contiguous
  v8f acc[2][2];
#pragma unroll
  for (int mi = 0; mi < 2; mi++)
#pragma unroll
    for (int ni = 0; ni < 2; ni++) acc[mi][ni] = zero8();
  for (int kb = 0; kb < 1024; kb += 32) {
    if (kb + 128 < 1024) {
      __builtin_prefetch(Ab + (long)(lane & 15) * 1024 + kb + 128, 0, 1);
      __builtin_prefetch(Wb + (long)(lane & 15) * 1024 + kb + 128, 0, 1);
    }
    v16bf a0 = load_a(Ab, 1024, kb, lane);
    v16bf a1 = load_a(Ab + (long)16 * 1024, 1024, kb, lane);
    v16bf b0 = load_b(Wb, 1024, kb, lane);
    v16bf b1 = load_b(Wb + (long)16 * 1024, 1024, kb, lane);
    acc[0][0] = wmma_bf16(a0, b0, acc[0][0]);
    acc[0][1] = wmma_bf16(a0, b1, acc[0][1]);
    acc[1][0] = wmma_bf16(a1, b0, acc[1][0]);
    acc[1][1] = wmma_bf16(a1, b1, acc[1][1]);
  }
  int col = lane & 15, half = lane >> 4;
#pragma unroll
  for (int mi = 0; mi < 2; mi++)
#pragma unroll
    for (int ni = 0; ni < 2; ni++) {
      int n = n0 + ni * 16 + col, h = n >> 7, p = n & 127;
#pragma unroll
      for (int r = 0; r < 8; r++) {
        int tok = m0 + mi * 16 + r + half * 8;
        int b = tok >> 11, s = tok & 2047;
        out[(((long)(b * 8 + h) * 2048 + s) * 128) + p] = (__bf16)(acc[mi][ni][r] * scale);
      }
    }
}

// ---------------- MoE v projection: vT[b,h,p,s] = sum_e g * (v_src @ V[h,e]), 32x32 tile ----------------
__global__ void vproj_kernel(const __bf16* __restrict__ vsrc, const __bf16* __restrict__ VT,
                             const float* __restrict__ gv, __bf16* __restrict__ vT) {
  int wid = (blockIdx.x * blockDim.x + threadIdx.x) >> 5;
  int lane = threadIdx.x & 31;
  int pt = wid & 3, h = (wid >> 2) & 7, tt = wid >> 5;  // 4 p-tiles(32), 8 heads, 128 token-tiles(32)
  int t0 = tt * 32, p0 = pt * 32;
  int col = lane & 15, half = lane >> 4;
  const __bf16* Ab = vsrc + (long)t0 * 1024;
  v8f acc[2][2];
#pragma unroll
  for (int mi = 0; mi < 2; mi++)
#pragma unroll
    for (int ni = 0; ni < 2; ni++) acc[mi][ni] = zero8();
  for (int e = 0; e < 8; e++) {
    float g[2][8];
    float gsum = 0.f;
#pragma unroll
    for (int mi = 0; mi < 2; mi++)
#pragma unroll
      for (int r = 0; r < 8; r++) {
        g[mi][r] = gv[((long)(t0 + mi * 16 + r + half * 8) * 8 + h) * 8 + e];
        gsum += g[mi][r];
      }
    gsum += __shfl_xor(gsum, 16);
    if (gsum == 0.f) continue;  // wave-uniform: expert unused by this token tile
    const __bf16* Bb = VT + ((long)(h * 8 + e) * 128 + p0) * 1024;  // VT[he][p][d]
    v8f t[2][2];
#pragma unroll
    for (int mi = 0; mi < 2; mi++)
#pragma unroll
      for (int ni = 0; ni < 2; ni++) t[mi][ni] = zero8();
    for (int kb = 0; kb < 1024; kb += 32) {
      v16bf a0 = load_a(Ab, 1024, kb, lane);
      v16bf a1 = load_a(Ab + (long)16 * 1024, 1024, kb, lane);
      v16bf b0 = load_b(Bb, 1024, kb, lane);
      v16bf b1 = load_b(Bb + (long)16 * 1024, 1024, kb, lane);
      t[0][0] = wmma_bf16(a0, b0, t[0][0]);
      t[0][1] = wmma_bf16(a0, b1, t[0][1]);
      t[1][0] = wmma_bf16(a1, b0, t[1][0]);
      t[1][1] = wmma_bf16(a1, b1, t[1][1]);
    }
#pragma unroll
    for (int mi = 0; mi < 2; mi++)
#pragma unroll
      for (int ni = 0; ni < 2; ni++)
#pragma unroll
        for (int r = 0; r < 8; r++) acc[mi][ni][r] += g[mi][r] * t[mi][ni][r];
  }
#pragma unroll
  for (int mi = 0; mi < 2; mi++)
#pragma unroll
    for (int ni = 0; ni < 2; ni++) {
      int p = p0 + ni * 16 + col;
      int tokbase = t0 + mi * 16 + half * 8;
      int b = tokbase >> 11, s = tokbase & 2047;
      v8bf pack;
#pragma unroll
      for (int r = 0; r < 8; r++) pack[r] = (__bf16)acc[mi][ni][r];
      *(v8bf*)(vT + ((long)(b * 8 + h) * 128 + p) * 2048 + s) = pack;
    }
}

// ---------------- flash causal attention: res[b,h,s,p] ----------------
__global__ void attn_kernel(const __bf16* __restrict__ qp, const __bf16* __restrict__ kp,
                            const __bf16* __restrict__ vT, __bf16* __restrict__ res) {
  __shared__ float pst[4][16][36];
  int widx = threadIdx.x >> 5;
  int lane = threadIdx.x & 31;
  int wid = blockIdx.x * 4 + widx;
  int qt = wid & 127, bh = wid >> 7;
  int q0 = qt * 16;
  const __bf16* qbase = qp + ((long)bh * 2048 + q0) * 128;
  const __bf16* kbase = kp + (long)bh * 2048 * 128;
  const __bf16* vbase = vT + (long)bh * 128 * 2048;
  int col = lane & 15, half = lane >> 4;

  v16bf aq[4];
#pragma unroll
  for (int j = 0; j < 4; j++) aq[j] = load_a(qbase, 128, j * 32, lane);

  v8f acc[8];
#pragma unroll
  for (int t = 0; t < 8; t++) acc[t] = zero8();
  float m_i[8], l_i[8];
#pragma unroll
  for (int r = 0; r < 8; r++) { m_i[r] = -1e30f; l_i[r] = 0.f; }

  int kend = q0 + 16;
  for (int k0 = 0; k0 < kend; k0 += 32) {
    // scores = q . k^T  (two 16-wide key groups)
    v8f sc[2];
#pragma unroll
    for (int n = 0; n < 2; n++) {
      v8f c = zero8();
#pragma unroll
      for (int j = 0; j < 4; j++) {
        v16bf b = load_b(kbase + (long)(k0 + n * 16) * 128, 128, j * 32, lane);
        c = wmma_bf16(aq[j], b, c);
      }
      sc[n] = c;
    }
    // causal mask + running softmax stats (row = r + half*8, col = key lane)
    float scale_old[8];
#pragma unroll
    for (int r = 0; r < 8; r++) {
      int row = r + half * 8;
#pragma unroll
      for (int n = 0; n < 2; n++) {
        int key = k0 + n * 16 + col;
        if (key > q0 + row) sc[n][r] = -1e30f;
      }
      float mx = fmaxf(sc[0][r], sc[1][r]);
      for (int m = 8; m >= 1; m >>= 1) mx = fmaxf(mx, __shfl_xor(mx, m));
      float mn = fmaxf(m_i[r], mx);
      scale_old[r] = __expf(m_i[r] - mn);
      m_i[r] = mn;
      float s0 = __expf(sc[0][r] - mn);
      float s1 = __expf(sc[1][r] - mn);
      sc[0][r] = s0; sc[1][r] = s1;
      float rs = s0 + s1;
      for (int m = 8; m >= 1; m >>= 1) rs += __shfl_xor(rs, m);
      l_i[r] = l_i[r] * scale_old[r] + rs;
    }
#pragma unroll
    for (int t = 0; t < 8; t++)
#pragma unroll
      for (int r = 0; r < 8; r++) acc[t][r] *= scale_old[r];

    // C-layout probs -> A-layout bf16 via per-wave LDS stage (DS in-order within a wave)
#pragma unroll
    for (int n = 0; n < 2; n++)
#pragma unroll
      for (int r = 0; r < 8; r++)
        pst[widx][r + half * 8][n * 16 + col] = sc[n][r];
    v16bf ap;
    {
      int row = lane & 15, kh = half * 8;
#pragma unroll
      for (int i = 0; i < 8; i++)  ap[i]     = (__bf16)pst[widx][row][kh + i];
#pragma unroll
      for (int i = 0; i < 8; i++)  ap[i + 8] = (__bf16)pst[widx][row][16 + kh + i];
    }
    // out += probs @ v   (v transposed: vT[p][s] so K contiguous)
#pragma unroll
    for (int t = 0; t < 8; t++) {
      v16bf bv = load_b(vbase + (long)(t * 16) * 2048 + k0, 2048, 0, lane);
      acc[t] = wmma_bf16(ap, bv, acc[t]);
    }
  }
  float inv[8];
#pragma unroll
  for (int r = 0; r < 8; r++) inv[r] = (l_i[r] > 0.f) ? 1.f / l_i[r] : 0.f;
#pragma unroll
  for (int t = 0; t < 8; t++) {
    int p = t * 16 + col;
#pragma unroll
    for (int r = 0; r < 8; r++)
      res[((long)bh * 2048 + q0 + r + half * 8) * 128 + p] = (__bf16)(acc[t][r] * inv[r]);
  }
}

// ---------------- MoE output projection: out[b,s,d] = sum_{h,e} g * (res @ O[h,e]), 32x32 tile ----------------
__global__ void oproj_kernel(const __bf16* __restrict__ res, const __bf16* __restrict__ OT,
                             const float* __restrict__ go, float* __restrict__ out) {
  int wid = (blockIdx.x * blockDim.x + threadIdx.x) >> 5;
  int lane = threadIdx.x & 31;
  int dt = wid & 31, tt = wid >> 5;   // 32 d-tiles(32), 128 token-tiles(32)
  int t0 = tt * 32, d0 = dt * 32;
  int col = lane & 15, half = lane >> 4;
  int b = t0 >> 11, s0 = t0 & 2047;
  v8f acc[2][2];
#pragma unroll
  for (int mi = 0; mi < 2; mi++)
#pragma unroll
    for (int ni = 0; ni < 2; ni++) acc[mi][ni] = zero8();
  for (int h = 0; h < 8; h++) {
    const __bf16* Ab = res + ((long)(b * 8 + h) * 2048 + s0) * 128;
    for (int e = 0; e < 8; e++) {
      float g[2][8];
      float gsum = 0.f;
#pragma unroll
      for (int mi = 0; mi < 2; mi++)
#pragma unroll
        for (int r = 0; r < 8; r++) {
          g[mi][r] = go[((long)(t0 + mi * 16 + r + half * 8) * 8 + h) * 8 + e];
          gsum += g[mi][r];
        }
      gsum += __shfl_xor(gsum, 16);
      if (gsum == 0.f) continue;
      const __bf16* Bb = OT + ((long)(h * 8 + e) * 1024 + d0) * 128;  // OT[he][d][p]
      v8f t[2][2];
#pragma unroll
      for (int mi = 0; mi < 2; mi++)
#pragma unroll
        for (int ni = 0; ni < 2; ni++) t[mi][ni] = zero8();
#pragma unroll
      for (int kb = 0; kb < 128; kb += 32) {
        v16bf a0 = load_a(Ab, 128, kb, lane);
        v16bf a1 = load_a(Ab + (long)16 * 128, 128, kb, lane);
        v16bf b0 = load_b(Bb, 128, kb, lane);
        v16bf b1 = load_b(Bb + (long)16 * 128, 128, kb, lane);
        t[0][0] = wmma_bf16(a0, b0, t[0][0]);
        t[0][1] = wmma_bf16(a0, b1, t[0][1]);
        t[1][0] = wmma_bf16(a1, b0, t[1][0]);
        t[1][1] = wmma_bf16(a1, b1, t[1][1]);
      }
#pragma unroll
      for (int mi = 0; mi < 2; mi++)
#pragma unroll
        for (int ni = 0; ni < 2; ni++)
#pragma unroll
          for (int r = 0; r < 8; r++) acc[mi][ni][r] += g[mi][r] * t[mi][ni][r];
    }
  }
#pragma unroll
  for (int mi = 0; mi < 2; mi++)
#pragma unroll
    for (int ni = 0; ni < 2; ni++) {
      int d = d0 + ni * 16 + col;
#pragma unroll
      for (int r = 0; r < 8; r++)
        out[(long)(t0 + mi * 16 + r + half * 8) * 1024 + d] = acc[mi][ni][r];
    }
}

// ---------------- launcher ----------------
extern "C" void kernel_launch(void* const* d_in, const int* in_sizes, int n_in,
                              void* d_out, int out_size, void* d_ws, size_t ws_size,
                              hipStream_t stream) {
  (void)in_sizes; (void)n_in; (void)out_size; (void)ws_size;
  const float* q_src = (const float*)d_in[0];
  const float* k_src = (const float*)d_in[1];
  const float* v_src = (const float*)d_in[2];
  const float* wq    = (const float*)d_in[3];
  const float* wk    = (const float*)d_in[4];
  const float* Vw    = (const float*)d_in[5];
  const float* Ow    = (const float*)d_in[6];
  const float* sel_v = (const float*)d_in[7];
  const float* sel_o = (const float*)d_in[8];
  float* out = (float*)d_out;

  char* ws = (char*)d_ws;
  size_t off = 0;
  auto alloc = [&](size_t bytes) -> void* {
    size_t o = (off + 255) & ~(size_t)255;
    off = o + bytes;
    return (void*)(ws + o);
  };
  const size_t BS = 4096;                  // B*S
  const size_t nsrc  = BS * 1024;          // B*S*D      = 4.19M
  const size_t nproj = nsrc;               // B*H*S*P    = 4.19M
  const size_t nexp  = 64ull * 128 * 1024; // H*E*P*D    = 8.39M
  __bf16* bq  = (__bf16*)alloc(nsrc * 2);
  __bf16* bk  = (__bf16*)alloc(nsrc * 2);
  __bf16* bv  = (__bf16*)alloc(nsrc * 2);
  __bf16* bwq = (__bf16*)alloc(1024 * 1024 * 2);
  __bf16* bwk = (__bf16*)alloc(1024 * 1024 * 2);
  __bf16* VT  = (__bf16*)alloc(nexp * 2);
  __bf16* OT  = (__bf16*)alloc(nexp * 2);
  __bf16* qp  = (__bf16*)alloc(nproj * 2);
  __bf16* kp  = (__bf16*)alloc(nproj * 2);
  __bf16* vT  = (__bf16*)alloc(nproj * 2);
  __bf16* rs  = (__bf16*)alloc(nproj * 2);
  float*  gv  = (float*)alloc(BS * 64 * 4);
  float*  go  = (float*)alloc(BS * 64 * 4);

  cvt_bf16_kernel<<<(int)((nsrc + 255) / 256), 256, 0, stream>>>(q_src, bq, (long)nsrc);
  cvt_bf16_kernel<<<(int)((nsrc + 255) / 256), 256, 0, stream>>>(k_src, bk, (long)nsrc);
  cvt_bf16_kernel<<<(int)((nsrc + 255) / 256), 256, 0, stream>>>(v_src, bv, (long)nsrc);
  cvt_bf16_kernel<<<4096, 256, 0, stream>>>(wq, bwq, 1024 * 1024);
  cvt_bf16_kernel<<<4096, 256, 0, stream>>>(wk, bwk, 1024 * 1024);

  dim3 tb(32, 8);
  tcvt_kernel<<<dim3(4, 32, 64), tb, 0, stream>>>(Vw, VT, 1024, 128);   // V[h,e,d,p] -> VT[he][p][d]
  tcvt_kernel<<<dim3(32, 4, 64), tb, 0, stream>>>(Ow, OT, 128, 1024);   // O[h,e,p,d] -> OT[he][d][p]

  gates_kernel<<<(int)BS, 64, 0, stream>>>(k_src, sel_v, gv);
  gates_kernel<<<(int)BS, 64, 0, stream>>>(q_src, sel_o, go);

  const float SCALE = 0.29730177875068026f;  // 128^-0.25
  qkproj_kernel<<<1024, 128, 0, stream>>>(bq, bwq, qp, SCALE);
  qkproj_kernel<<<1024, 128, 0, stream>>>(bk, bwk, kp, SCALE);
  vproj_kernel<<<1024, 128, 0, stream>>>(bv, VT, gv, vT);
  attn_kernel<<<512, 128, 0, stream>>>(qp, kp, vT, rs);
  oproj_kernel<<<1024, 128, 0, stream>>>(rs, OT, go, out);
}
